// AttnDecoderRNN_70317204570616
// MI455X (gfx1250) — matrix operand order repaired and checked by reading, not compiled, not run on hardware
//
#include <hip/hip_runtime.h>
#include <math.h>

// Problem constants (from the reference)
#define B_  64
#define E_  1024
#define H_  1024
#define L_  512
#define V_  50257
#define EH_ 2048      // E + H
#define H3_ 3072      // 3 * H

typedef __attribute__((ext_vector_type(16))) __bf16 v16bf;
typedef __attribute__((ext_vector_type(2)))  __bf16 v2bf;
typedef __attribute__((ext_vector_type(8)))  float  v8f;

// ---- f32 -> bf16 conversion helpers -------------------------------------
// (__bf16) casts lower to native v_cvt_pk_bf16_f32 on gfx1250 (verified in
// round-3 disassembly); keep the builtin path first in case it appears.
__device__ __forceinline__ unsigned pack2(float lo, float hi) {
#if __has_builtin(__builtin_amdgcn_cvt_pk_bf16_f32)
  auto r = __builtin_amdgcn_cvt_pk_bf16_f32(lo, hi);
  return __builtin_bit_cast(unsigned, r);
#else
  v2bf r;
  r[0] = (__bf16)lo;
  r[1] = (__bf16)hi;
  return __builtin_bit_cast(unsigned, r);
#endif
}
__device__ __forceinline__ unsigned short f2bfs(float f) {
  __bf16 b = (__bf16)f;
  return __builtin_bit_cast(unsigned short, b);
}

union BfVec { v16bf v; unsigned u[8]; };

// ---------------------------------------------------------------------------
// C[64,N] = act( A[64,K] @ W[N,K]^T + bias ),  REQUIRES K % 64 == 0.
// A is pre-converted packed bf16 (loaded raw, no VALU conversion).
// W is streamed f32 from HBM with a 2-deep ping-pong register pipeline
// (unroll-by-2 -> no cross-iteration register moves) and converted with
// native v_cvt_pk_bf16_f32 (8 per k-step).
// One wave owns a 16-column strip and all 4 M-tiles (M=64).
// Wave32 VGPR layouts per CDNA5 ISA 7.12.2, but A/B use contiguous k = 16*(lane/16) + s.
// ---------------------------------------------------------------------------
__global__ __launch_bounds__(256) void gemm_bf16_wmma(
    const unsigned short* __restrict__ Abf, // [64, K] bf16, row-major
    const float* __restrict__ W,            // [N, K]  f32,  row-major
    const float* __restrict__ bias,         // [N] or nullptr
    float* __restrict__ outF,               // [64, N] f32 or nullptr
    unsigned short* __restrict__ outBf,     // [64, N] bf16 or nullptr
    int N, int K, int relu)
{
  const int wave   = threadIdx.x >> 5;
  const int lane   = threadIdx.x & 31;
  const int nTiles = (N + 15) >> 4;
  const int nTile  = blockIdx.x * 8 + wave;
  if (nTile >= nTiles) return;           // wave-uniform exit: EXEC stays full

  const int lh = lane >> 4;              // lane-half (K split)
  const int nn = lane & 15;              // col within tile / row within tile
  const int n  = nTile * 16 + nn;
  const int nc = (n < N) ? n : (N - 1);  // clamp loads; stores masked below
  const float* __restrict__ wrow = W + (size_t)nc * K;

  const unsigned* __restrict__ Au = (const unsigned*)Abf;  // 2 bf16 / uint
  const int Ku = K >> 1;

  v8f acc0 = {}, acc1 = {}, acc2 = {}, acc3 = {};

  // ---- preload first B (weight) block: 16 consecutive K of column n ----
  float4 b0, b1, b2, b3;      // ping buffer (even k-steps)
  float4 c0, c1, c2, c3;      // pong buffer (odd  k-steps)
  {
    const float4* bp = (const float4*)(wrow + lh * 16);
    b0 = bp[0]; b1 = bp[1]; b2 = bp[2]; b3 = bp[3];
  }

// convert a float4 quad into the 32-K bf16 B-tile and run the 4 WMMAs
#define DO_KSTEP(X0, X1, X2, X3, KOFF) {                                      \
    const unsigned* abase = Au + (size_t)nn * Ku + ((KOFF) >> 1) + lh * 8;    \
    uint4 a0l = *(const uint4*)(abase + 0 * 16 * Ku);                         \
    uint4 a0h = *(const uint4*)(abase + 0 * 16 * Ku + 4);                     \
    uint4 a1l = *(const uint4*)(abase + 1 * 16 * Ku);                         \
    uint4 a1h = *(const uint4*)(abase + 1 * 16 * Ku + 4);                     \
    uint4 a2l = *(const uint4*)(abase + 2 * 16 * Ku);                         \
    uint4 a2h = *(const uint4*)(abase + 2 * 16 * Ku + 4);                     \
    uint4 a3l = *(const uint4*)(abase + 3 * 16 * Ku);                         \
    uint4 a3h = *(const uint4*)(abase + 3 * 16 * Ku + 4);                     \
    BfVec bm;                                                                 \
    bm.u[0] = pack2(X0.x, X0.y); bm.u[1] = pack2(X0.z, X0.w);                 \
    bm.u[2] = pack2(X1.x, X1.y); bm.u[3] = pack2(X1.z, X1.w);                 \
    bm.u[4] = pack2(X2.x, X2.y); bm.u[5] = pack2(X2.z, X2.w);                 \
    bm.u[6] = pack2(X3.x, X3.y); bm.u[7] = pack2(X3.z, X3.w);                 \
    BfVec am;                                                                 \
    am.u[0] = a0l.x; am.u[1] = a0l.y; am.u[2] = a0l.z; am.u[3] = a0l.w;       \
    am.u[4] = a0h.x; am.u[5] = a0h.y; am.u[6] = a0h.z; am.u[7] = a0h.w;       \
    acc0 = __builtin_amdgcn_wmma_f32_16x16x32_bf16(                           \
        false, am.v, false, bm.v, (short)0, acc0, false, false);              \
    am.u[0] = a1l.x; am.u[1] = a1l.y; am.u[2] = a1l.z; am.u[3] = a1l.w;       \
    am.u[4] = a1h.x; am.u[5] = a1h.y; am.u[6] = a1h.z; am.u[7] = a1h.w;       \
    acc1 = __builtin_amdgcn_wmma_f32_16x16x32_bf16(                           \
        false, am.v, false, bm.v, (short)0, acc1, false, false);              \
    am.u[0] = a2l.x; am.u[1] = a2l.y; am.u[2] = a2l.z; am.u[3] = a2l.w;       \
    am.u[4] = a2h.x; am.u[5] = a2h.y; am.u[6] = a2h.z; am.u[7] = a2h.w;       \
    acc2 = __builtin_amdgcn_wmma_f32_16x16x32_bf16(                           \
        false, am.v, false, bm.v, (short)0, acc2, false, false);              \
    am.u[0] = a3l.x; am.u[1] = a3l.y; am.u[2] = a3l.z; am.u[3] = a3l.w;       \
    am.u[4] = a3h.x; am.u[5] = a3h.y; am.u[6] = a3h.z; am.u[7] = a3h.w;       \
    acc3 = __builtin_amdgcn_wmma_f32_16x16x32_bf16(                           \
        false, am.v, false, bm.v, (short)0, acc3, false, false); }

  for (int k0 = 0; k0 < K; k0 += 64) {
    // ---- phase A: consume ping (k0), prefetch pong (k0+32) ----
    {
      const float4* bp = (const float4*)(wrow + (k0 + 32) + lh * 16);
      c0 = bp[0]; c1 = bp[1]; c2 = bp[2]; c3 = bp[3];
      __builtin_prefetch(wrow + (k0 + 96) + lh * 16, 0, 1);
    }
    DO_KSTEP(b0, b1, b2, b3, k0)

    // ---- phase B: consume pong (k0+32), prefetch ping (k0+64) ----
    if (k0 + 64 < K) {
      const float4* bp = (const float4*)(wrow + (k0 + 64) + lh * 16);
      b0 = bp[0]; b1 = bp[1]; b2 = bp[2]; b3 = bp[3];
    }
    DO_KSTEP(c0, c1, c2, c3, k0 + 32)
  }
#undef DO_KSTEP

  const float bv = bias ? bias[nc] : 0.0f;
  if (n < N) {
#define MT_STORE(MT, ACC) {                                                   \
    _Pragma("unroll")                                                         \
    for (int i = 0; i < 8; ++i) {                                             \
      int m = (MT) * 16 + i + lh * 8;                                         \
      float v = ACC[i] + bv;                                                  \
      if (relu) v = fmaxf(v, 0.0f);                                           \
      if (outF)  outF[(size_t)m * N + n]  = v;                                \
      if (outBf) outBf[(size_t)m * N + n] = f2bfs(v);                         \
    } }
    MT_STORE(0, acc0)
    MT_STORE(1, acc1)
    MT_STORE(2, acc2)
    MT_STORE(3, acc3)
#undef MT_STORE
  }
}

// ---------------------------------------------------------------------------
// Embedding gather + build bf16 GEMM inputs:
//   cat1_bf = [emb | hidden], cat2_bf[:, :E] = emb, hid_bf = hidden
// ---------------------------------------------------------------------------
__global__ __launch_bounds__(256) void k_embed_cat(
    const int* __restrict__ ids, const float* __restrict__ embW,
    const float* __restrict__ hidden,
    unsigned short* __restrict__ cat1bf, unsigned short* __restrict__ cat2bf,
    unsigned short* __restrict__ hidbf)
{
  const int b = blockIdx.x;
  const float* er = embW + (size_t)ids[b] * E_;
  for (int e = threadIdx.x; e < E_; e += blockDim.x) {
    unsigned short us = f2bfs(er[e]);
    cat1bf[(size_t)b * EH_ + e] = us;
    cat2bf[(size_t)b * EH_ + e] = us;
  }
  for (int j = threadIdx.x; j < H_; j += blockDim.x) {
    unsigned short us = f2bfs(hidden[(size_t)b * H_ + j]);
    cat1bf[(size_t)b * EH_ + E_ + j] = us;
    hidbf[(size_t)b * H_ + j] = us;
  }
}

// ---------------------------------------------------------------------------
// In-place row softmax (rows of length Lr), one block per row, 256 threads
// ---------------------------------------------------------------------------
__global__ __launch_bounds__(256) void k_softmax_rows(float* __restrict__ x, int Lr)
{
  __shared__ float red[256];
  const int tid = threadIdx.x;
  float* xr = x + (size_t)blockIdx.x * Lr;

  float m = -3.4e38f;
  for (int i = tid; i < Lr; i += 256) m = fmaxf(m, xr[i]);
  red[tid] = m; __syncthreads();
  for (int s = 128; s > 0; s >>= 1) {
    if (tid < s) red[tid] = fmaxf(red[tid], red[tid + s]);
    __syncthreads();
  }
  const float mx = red[0]; __syncthreads();

  float sum = 0.0f;
  for (int i = tid; i < Lr; i += 256) sum += __expf(xr[i] - mx);
  red[tid] = sum; __syncthreads();
  for (int s = 128; s > 0; s >>= 1) {
    if (tid < s) red[tid] += red[tid + s];
    __syncthreads();
  }
  const float inv = 1.0f / red[0];

  for (int i = tid; i < Lr; i += 256) xr[i] = __expf(xr[i] - mx) * inv;
}

// ---------------------------------------------------------------------------
// attn_applied[b,d] = sum_l w[b,l] * enc[b,l,d]  -> cat2_bf[b, E + d]
// One block per batch row, 1024 threads, coalesced over d.
// ---------------------------------------------------------------------------
__global__ __launch_bounds__(1024) void k_attn_apply(
    const float* __restrict__ w, const float* __restrict__ enc,
    unsigned short* __restrict__ cat2bf)
{
  const int b = blockIdx.x;
  const int d = threadIdx.x;
  const float* wb = w + (size_t)b * L_;
  const float* eb = enc + (size_t)b * L_ * H_;
  float s = 0.0f;
  for (int l = 0; l < L_; ++l)
    s = fmaf(wb[l], eb[(size_t)l * H_ + d], s);
  cat2bf[(size_t)b * EH_ + E_ + d] = f2bfs(s);
}

// ---------------------------------------------------------------------------
// GRU gate math (torch gate order r,z,n); biases already folded in by GEMMs.
// Writes new_hidden in f32 (tuple output) and bf16 (vocab-GEMM input).
// ---------------------------------------------------------------------------
__global__ __launch_bounds__(256) void k_gru(
    const float* __restrict__ gi, const float* __restrict__ gh,
    const float* __restrict__ hidden,
    float* __restrict__ nhF, unsigned short* __restrict__ nhBf)
{
  const int idx = blockIdx.x * 256 + threadIdx.x;
  if (idx >= B_ * H_) return;
  const int b = idx >> 10;
  const int j = idx & (H_ - 1);
  const float* gib = gi + (size_t)b * H3_;
  const float* ghb = gh + (size_t)b * H3_;
  const float r = 1.0f / (1.0f + __expf(-(gib[j]          + ghb[j])));
  const float z = 1.0f / (1.0f + __expf(-(gib[H_ + j]     + ghb[H_ + j])));
  const float n = tanhf(gib[2 * H_ + j] + r * ghb[2 * H_ + j]);
  const float h = hidden[(size_t)b * H_ + j];
  const float nh = (1.0f - z) * n + z * h;
  nhF[idx]  = nh;
  nhBf[idx] = f2bfs(nh);
}

// ---------------------------------------------------------------------------
// In-place row log-softmax (rows of length Lr=50257), 1024 threads per row
// ---------------------------------------------------------------------------
__global__ __launch_bounds__(1024) void k_logsoftmax_rows(float* __restrict__ x, int Lr)
{
  __shared__ float red[1024];
  const int tid = threadIdx.x;
  float* xr = x + (size_t)blockIdx.x * Lr;

  float m = -3.4e38f;
  for (int i = tid; i < Lr; i += 1024) m = fmaxf(m, xr[i]);
  red[tid] = m; __syncthreads();
  for (int s = 512; s > 0; s >>= 1) {
    if (tid < s) red[tid] = fmaxf(red[tid], red[tid + s]);
    __syncthreads();
  }
  const float mx = red[0]; __syncthreads();

  float sum = 0.0f;
  for (int i = tid; i < Lr; i += 1024) sum += __expf(xr[i] - mx);
  red[tid] = sum; __syncthreads();
  for (int s = 512; s > 0; s >>= 1) {
    if (tid < s) red[tid] += red[tid + s];
    __syncthreads();
  }
  const float lse = mx + logf(red[0]);

  for (int i = tid; i < Lr; i += 1024) xr[i] = xr[i] - lse;
}

// ---------------------------------------------------------------------------
extern "C" void kernel_launch(void* const* d_in, const int* in_sizes, int n_in,
                              void* d_out, int out_size, void* d_ws, size_t ws_size,
                              hipStream_t stream)
{
  const int*   input_ids = (const int*)  d_in[0];
  const float* hidden    = (const float*)d_in[1];
  const float* enc       = (const float*)d_in[2];
  const float* emb_W     = (const float*)d_in[3];
  const float* attn_W    = (const float*)d_in[4];
  const float* attn_b    = (const float*)d_in[5];
  const float* comb_W    = (const float*)d_in[6];
  const float* comb_b    = (const float*)d_in[7];
  const float* gru_Wih   = (const float*)d_in[8];
  const float* gru_Whh   = (const float*)d_in[9];
  const float* gru_bih   = (const float*)d_in[10];
  const float* gru_bhh   = (const float*)d_in[11];
  const float* out_W     = (const float*)d_in[12];
  const float* out_b     = (const float*)d_in[13];

  float* out_logits = (float*)d_out;                      // [B, V]
  float* new_hidden = out_logits + (size_t)B_ * V_;       // [B, H]

  // Workspace layout (byte-based carve-up)
  char* wsb = (char*)d_ws;
  auto take = [&](size_t bytes) { char* p = wsb; wsb += (bytes + 255) & ~(size_t)255; return p; };
  unsigned short* cat1bf = (unsigned short*)take((size_t)B_ * EH_ * 2);
  unsigned short* cat2bf = (unsigned short*)take((size_t)B_ * EH_ * 2);
  unsigned short* hidbf  = (unsigned short*)take((size_t)B_ * H_  * 2);
  unsigned short* xbf    = (unsigned short*)take((size_t)B_ * E_  * 2);
  unsigned short* nhbf   = (unsigned short*)take((size_t)B_ * H_  * 2);
  float*          attw   = (float*)take((size_t)B_ * L_  * 4);
  float*          gi     = (float*)take((size_t)B_ * H3_ * 4);
  float*          gh     = (float*)take((size_t)B_ * H3_ * 4);

  auto gemm_grid = [](int N) { return (((N + 15) >> 4) + 7) / 8; };

  // 1) embedding gather + bf16 concat inputs
  k_embed_cat<<<B_, 256, 0, stream>>>(input_ids, emb_W, hidden, cat1bf, cat2bf, hidbf);

  // 2) attention logits: [B,L] = cat1 @ attn_W^T + attn_b ; softmax rows
  gemm_bf16_wmma<<<gemm_grid(L_), 256, 0, stream>>>(cat1bf, attn_W, attn_b,
                                                    attw, nullptr, L_, EH_, 0);
  k_softmax_rows<<<B_, 256, 0, stream>>>(attw, L_);

  // 3) attn_applied -> cat2_bf[:, E:]
  k_attn_apply<<<B_, 1024, 0, stream>>>(attw, enc, cat2bf);

  // 4) x = relu(cat2 @ comb_W^T + comb_b), emitted directly as bf16
  gemm_bf16_wmma<<<gemm_grid(E_), 256, 0, stream>>>(cat2bf, comb_W, comb_b,
                                                    nullptr, xbf, E_, EH_, 1);

  // 5) GRU input/hidden projections (independent)
  gemm_bf16_wmma<<<gemm_grid(H3_), 256, 0, stream>>>(xbf, gru_Wih, gru_bih,
                                                     gi, nullptr, H3_, E_, 0);
  gemm_bf16_wmma<<<gemm_grid(H3_), 256, 0, stream>>>(hidbf, gru_Whh, gru_bhh,
                                                     gh, nullptr, H3_, H_, 0);

  // 6) gate math -> new_hidden (f32 tuple output + bf16 GEMM input)
  k_gru<<<(B_ * H_) / 256, 256, 0, stream>>>(gi, gh, hidden, new_hidden, nhbf);

  // 7) vocab projection (dominant GEMM, HBM-bound on out_W stream)
  gemm_bf16_wmma<<<gemm_grid(V_), 256, 0, stream>>>(nhbf, out_W, out_b,
                                                    out_logits, nullptr, V_, H_, 0);

  // 8) log-softmax rows in place
  k_logsoftmax_rows<<<B_, 1024, 0, stream>>>(out_logits, V_);
}
